// VisionAttention_48069273977252
// MI455X (gfx1250) — matrix operand order repaired and checked by reading, not compiled, not run on hardware
//
#include <hip/hip_runtime.h>

#define SEQ   4096
#define EMB   1280
#define HEADS 16
#define HD    80
#define QKVN  3840

typedef __attribute__((ext_vector_type(16))) __bf16       v16bf;
typedef __attribute__((ext_vector_type(8)))  float        v8f;
typedef __attribute__((ext_vector_type(4)))  unsigned int u32x4;
typedef __attribute__((ext_vector_type(4)))  int          v4i;

union ABFrag { v16bf v; u32x4 q[2]; unsigned short s[16]; };

#if defined(__has_builtin)
#if __has_builtin(__builtin_amdgcn_global_load_async_to_lds_b128) && \
    __has_builtin(__builtin_amdgcn_s_wait_asynccnt)
#define USE_ASYNC_LDS 1
#endif
#endif

#ifdef USE_ASYNC_LDS
typedef __attribute__((address_space(1))) v4i* gv4ip;
typedef __attribute__((address_space(3))) v4i* lv4ip;
#endif

__device__ __forceinline__ unsigned short f2bf(float f) {
  unsigned int u = __builtin_bit_cast(unsigned int, f);
  unsigned int r = u + 0x7FFFu + ((u >> 16) & 1u);
  return (unsigned short)(r >> 16);
}

// ---------------------------------------------------------------- f32 -> bf16
__global__ __launch_bounds__(256) void cvt_bf16_kernel(const float* __restrict__ s,
                                                       unsigned short* __restrict__ d,
                                                       int n4) {
  int i = blockIdx.x * 256 + threadIdx.x;
  if (i < n4) {
    float4 f = ((const float4*)s)[i];
    union { unsigned short u[4]; uint2 v; } o;
    o.u[0] = f2bf(f.x); o.u[1] = f2bf(f.y); o.u[2] = f2bf(f.z); o.u[3] = f2bf(f.w);
    ((uint2*)d)[i] = o.v;
  }
}

// ------------------------------------------------------- bf16 WMMA GEMM 64x128
// C[M,N] = A[M,K] * B[K,N] + bias ; A,B bf16 row-major; out bf16 or f32.
template <bool BF16OUT>
__global__ __launch_bounds__(256) void gemm_wmma_kernel(const unsigned short* __restrict__ A,
                                                        const unsigned short* __restrict__ B,
                                                        const float* __restrict__ bias,
                                                        void* __restrict__ Cv,
                                                        int M, int N, int K) {
  __shared__ __align__(16) unsigned short As[64][40];   // 64 x 32 (+8 pad)
  __shared__ __align__(16) unsigned short Bt[128][40];  // transposed: [n][k]

  const int tid = threadIdx.x;
  const int l   = tid & 31;
  const int w   = tid >> 5;
  const int lm  = l & 15;
  const int lh  = l >> 4;
  const int mw  = (w & 3) * 16;   // wave's 16-row strip inside 64
  const int nw  = (w >> 2) * 64;  // wave's 64-col strip inside 128

  const int m0 = blockIdx.y * 64;
  const int n0 = blockIdx.x * 128;

  v8f acc[4];
#pragma unroll
  for (int t = 0; t < 4; ++t) acc[t] = (v8f){0,0,0,0,0,0,0,0};

  // per-thread load coords
  const int arow = tid >> 2, akoff = (tid & 3) * 8;    // A: 64 rows x 32 k
  const int bk   = tid >> 3, bnoff = (tid & 7) * 16;   // B: 32 k x 128 n

  for (int k0 = 0; k0 < K; k0 += 32) {
    // stage A tile (row-major) -- async direct global->LDS when available
#ifdef USE_ASYNC_LDS
    __builtin_amdgcn_global_load_async_to_lds_b128(
        (gv4ip)(A + (size_t)(m0 + arow) * K + k0 + akoff),
        (lv4ip)&As[arow][akoff], 0, 0);
#else
    *(u32x4*)&As[arow][akoff] =
        *(const u32x4*)(A + (size_t)(m0 + arow) * K + k0 + akoff);
#endif
    // stage B tile transposed: Bt[n][k]
    {
      const unsigned short* bp = B + (size_t)(k0 + bk) * N + n0 + bnoff;
      union { u32x4 q; unsigned short s[8]; } t0, t1;
      t0.q = *(const u32x4*)bp;
      t1.q = *(const u32x4*)(bp + 8);
#pragma unroll
      for (int j = 0; j < 8; ++j) Bt[bnoff + j][bk] = t0.s[j];
#pragma unroll
      for (int j = 0; j < 8; ++j) Bt[bnoff + 8 + j][bk] = t1.s[j];
    }
#ifdef USE_ASYNC_LDS
    __builtin_amdgcn_s_wait_asynccnt(0);
#endif
    __syncthreads();

    ABFrag a;
    a.q[0] = *(const u32x4*)&As[mw + lm][lh * 16];
    a.q[1] = *(const u32x4*)&As[mw + lm][lh * 16 + 8];
#pragma unroll
    for (int t = 0; t < 4; ++t) {
      ABFrag b;
      const unsigned short* bp = &Bt[nw + 16 * t + lm][lh * 16];
      b.q[0] = *(const u32x4*)bp;
      b.q[1] = *(const u32x4*)(bp + 8);
      acc[t] = __builtin_amdgcn_wmma_f32_16x16x32_bf16(
          false, a.v, false, b.v, (short)0, acc[t], false, false);
    }
    __syncthreads();
  }

  // epilogue: C layout -> rows r+8*lh, col = lane n
#pragma unroll
  for (int t = 0; t < 4; ++t) {
    const int col = n0 + nw + 16 * t + lm;
    const float bv = bias[col];
#pragma unroll
    for (int r = 0; r < 8; ++r) {
      const int row = m0 + mw + 8 * lh + r;
      const float v = acc[t][r] + bv;
      if (BF16OUT)
        ((unsigned short*)Cv)[(size_t)row * N + col] = f2bf(v);
      else
        ((float*)Cv)[(size_t)row * N + col] = v;
    }
  }
}

// --------------------------------------------------- flash attention, 1 wave/block
// grid = (SEQ/16 q-tiles, HEADS). qkv bf16 [SEQ][3840], out bf16 [SEQ][1280].
__global__ __launch_bounds__(32) void attn_wmma_kernel(const unsigned short* __restrict__ qkv,
                                                       const int* __restrict__ cu,
                                                       int nseg,
                                                       unsigned short* __restrict__ outb) {
  __shared__ int rs[16], re[16];
  __shared__ __align__(16) unsigned short P[16][40];   // 16 x 32 probs (+pad)
  __shared__ __align__(16) unsigned short Vt[80][40];  // V transposed [dim][tok]

  const int l  = threadIdx.x;
  const int lm = l & 15;
  const int lh = l >> 4;
  const int qt = blockIdx.x;
  const int h  = blockIdx.y;

  if (l < 16) {
    const int row = qt * 16 + l;
    int s0 = 0, e0 = SEQ;
    for (int i = 0; i < nseg; ++i)
      if (row >= cu[i] && row < cu[i + 1]) { s0 = cu[i]; e0 = cu[i + 1]; }
    rs[l] = s0; re[l] = e0;
  }
  asm volatile("s_wait_dscnt 0" ::: "memory");

  int myrs[8], myre[8];
#pragma unroll
  for (int r = 0; r < 8; ++r) { myrs[r] = rs[8 * lh + r]; myre[r] = re[8 * lh + r]; }
  const int klo   = rs[0] & ~31;
  const int khi   = re[15];
  const int rsmax = rs[15];   // rows sorted: max segment start
  const int remin = re[0];    // min segment end

  // Q fragments: head dim 80 padded to 96 (3 K-steps of 32)
  ABFrag qf[3];
  {
    const unsigned short* qbase = qkv + (size_t)(qt * 16 + lm) * QKVN + h * HD;
    const u32x4 z = {0, 0, 0, 0};
#pragma unroll
    for (int ks = 0; ks < 3; ++ks) {
      const int d0 = 32 * ks + lh * 16;
      const int d1 = 32 * ks + lh * 16 + 8;
      qf[ks].q[0] = (d0 >= HD) ? z : *(const u32x4*)(qbase + d0);
      qf[ks].q[1] = (d1 >= HD) ? z : *(const u32x4*)(qbase + d1);
    }
  }

  float mrow[8], lrow[8];
  v8f o[5];
#pragma unroll
  for (int r = 0; r < 8; ++r) { mrow[r] = -3.0e38f; lrow[r] = 0.f; }
#pragma unroll
  for (int t = 0; t < 5; ++t) o[t] = (v8f){0,0,0,0,0,0,0,0};

  const float scale = 0.1118033988749895f;  // 1/sqrt(80)

  for (int k0 = klo; k0 < khi; k0 += 32) {
    // stage V tile transposed: lane = token
    {
      const unsigned short* vrow =
          qkv + (size_t)(k0 + l) * QKVN + 2 * EMB + h * HD;
#pragma unroll
      for (int j = 0; j < 10; ++j) {
        union { u32x4 q; unsigned short s[8]; } tmp;
        tmp.q = *(const u32x4*)(vrow + 8 * j);
#pragma unroll
        for (int e = 0; e < 8; ++e) Vt[8 * j + e][l] = tmp.s[e];
      }
    }

    // S = Q * K^T for two 16-token column tiles
    v8f s0v = (v8f){0,0,0,0,0,0,0,0};
    v8f s1v = (v8f){0,0,0,0,0,0,0,0};
    {
      const unsigned short* kbase = qkv + EMB + h * HD;
      const u32x4 z = {0, 0, 0, 0};
#pragma unroll
      for (int ks = 0; ks < 3; ++ks) {
        const int dk = 32 * ks + lh * 16;
        const size_t r0 = (size_t)(k0 + lm) * QKVN;
        const size_t r1 = (size_t)(k0 + 16 + lm) * QKVN;
        ABFrag b0, b1;
        b0.q[0] = (dk     >= HD) ? z : *(const u32x4*)(kbase + r0 + dk);
        b0.q[1] = (dk + 8 >= HD) ? z : *(const u32x4*)(kbase + r0 + dk + 8);
        b1.q[0] = (dk     >= HD) ? z : *(const u32x4*)(kbase + r1 + dk);
        b1.q[1] = (dk + 8 >= HD) ? z : *(const u32x4*)(kbase + r1 + dk + 8);
        s0v = __builtin_amdgcn_wmma_f32_16x16x32_bf16(
            false, qf[ks].v, false, b0.v, (short)0, s0v, false, false);
        s1v = __builtin_amdgcn_wmma_f32_16x16x32_bf16(
            false, qf[ks].v, false, b1.v, (short)0, s1v, false, false);
      }
    }

    // online softmax (per row r of this wave-half)
    float alpha[8];
    const bool interior = (k0 >= rsmax) && (k0 + 32 <= remin);
    if (interior) {
      // fast path: no masking work at all
#pragma unroll
      for (int r = 0; r < 8; ++r) {
        const float v0 = s0v[r] * scale;
        const float v1 = s1v[r] * scale;
        float t = fmaxf(v0, v1);
#pragma unroll
        for (int sh = 1; sh < 16; sh <<= 1) t = fmaxf(t, __shfl_xor(t, sh, 32));
        const float nm = fmaxf(mrow[r], t);
        alpha[r] = __expf(mrow[r] - nm);
        mrow[r] = nm;
        const float p0 = __expf(v0 - nm);
        const float p1 = __expf(v1 - nm);
        float ps = p0 + p1;
#pragma unroll
        for (int sh = 1; sh < 16; sh <<= 1) ps += __shfl_xor(ps, sh, 32);
        lrow[r] = alpha[r] * lrow[r] + ps;
        P[8 * lh + r][lm]      = f2bf(p0);
        P[8 * lh + r][16 + lm] = f2bf(p1);
      }
    } else {
      // boundary path: per-row segment mask
#pragma unroll
      for (int r = 0; r < 8; ++r) {
        const int c0 = k0 + lm;
        const int c1 = k0 + 16 + lm;
        const bool ok0 = (c0 >= myrs[r]) && (c0 < myre[r]);
        const bool ok1 = (c1 >= myrs[r]) && (c1 < myre[r]);
        const float v0 = ok0 ? s0v[r] * scale : -3.0e38f;
        const float v1 = ok1 ? s1v[r] * scale : -3.0e38f;
        float t = fmaxf(v0, v1);
#pragma unroll
        for (int sh = 1; sh < 16; sh <<= 1) t = fmaxf(t, __shfl_xor(t, sh, 32));
        const float nm = fmaxf(mrow[r], t);
        alpha[r] = __expf(mrow[r] - nm);
        mrow[r] = nm;
        const float p0 = ok0 ? __expf(v0 - nm) : 0.f;
        const float p1 = ok1 ? __expf(v1 - nm) : 0.f;
        float ps = p0 + p1;
#pragma unroll
        for (int sh = 1; sh < 16; sh <<= 1) ps += __shfl_xor(ps, sh, 32);
        lrow[r] = alpha[r] * lrow[r] + ps;
        P[8 * lh + r][lm]      = f2bf(p0);
        P[8 * lh + r][16 + lm] = f2bf(p1);
      }
    }
    asm volatile("s_wait_dscnt 0" ::: "memory");

    // O = alpha*O + P*V
    ABFrag pa;
    pa.q[0] = *(const u32x4*)&P[lm][lh * 16];
    pa.q[1] = *(const u32x4*)&P[lm][lh * 16 + 8];
#pragma unroll
    for (int t = 0; t < 5; ++t) {
      ABFrag vb;
      const unsigned short* vp = &Vt[16 * t + lm][lh * 16];
      vb.q[0] = *(const u32x4*)vp;
      vb.q[1] = *(const u32x4*)(vp + 8);
      v8f oo = o[t];
#pragma unroll
      for (int r = 0; r < 8; ++r) oo[r] *= alpha[r];
      o[t] = __builtin_amdgcn_wmma_f32_16x16x32_bf16(
          false, pa.v, false, vb.v, (short)0, oo, false, false);
    }
    asm volatile("" ::: "memory");
  }

  // normalize + store
  float inv[8];
#pragma unroll
  for (int r = 0; r < 8; ++r) inv[r] = 1.f / lrow[r];
#pragma unroll
  for (int t = 0; t < 5; ++t) {
#pragma unroll
    for (int r = 0; r < 8; ++r) {
      const int row = qt * 16 + 8 * lh + r;
      const int col = h * HD + 16 * t + lm;
      outb[(size_t)row * EMB + col] = f2bf(o[t][r] * inv[r]);
    }
  }
}

// ---------------------------------------------------------------------- launch
extern "C" void kernel_launch(void* const* d_in, const int* in_sizes, int n_in,
                              void* d_out, int out_size, void* d_ws, size_t ws_size,
                              hipStream_t stream) {
  const float* x     = (const float*)d_in[0];
  const float* w_qkv = (const float*)d_in[1];
  const float* b_qkv = (const float*)d_in[2];
  const float* w_out = (const float*)d_in[3];
  const float* b_out = (const float*)d_in[4];
  const int*   cu    = (const int*)d_in[5];
  const int nseg = in_sizes[5] - 1;

  char* ws = (char*)d_ws;
  unsigned short* xb    = (unsigned short*)(ws);
  unsigned short* wqkvb = (unsigned short*)(ws + 10485760);          // 4096*1280*2
  unsigned short* wob   = (unsigned short*)(ws + 10485760 + 9830400);
  unsigned short* qkvb  = (unsigned short*)(ws + 23592960);
  unsigned short* attnb = (unsigned short*)(ws + 23592960 + 31457280);

  const int nx = SEQ * EMB / 4, nwq = EMB * QKVN / 4, nwo = EMB * EMB / 4;
  cvt_bf16_kernel<<<(nx  + 255) / 256, 256, 0, stream>>>(x,     xb,    nx);
  cvt_bf16_kernel<<<(nwq + 255) / 256, 256, 0, stream>>>(w_qkv, wqkvb, nwq);
  cvt_bf16_kernel<<<(nwo + 255) / 256, 256, 0, stream>>>(w_out, wob,   nwo);

  gemm_wmma_kernel<true><<<dim3(QKVN / 128, SEQ / 64), 256, 0, stream>>>(
      xb, wqkvb, b_qkv, qkvb, SEQ, QKVN, EMB);

  attn_wmma_kernel<<<dim3(SEQ / 16, HEADS), 32, 0, stream>>>(qkvb, cu, nseg, attnb);

  gemm_wmma_kernel<false><<<dim3(EMB / 128, SEQ / 64), 256, 0, stream>>>(
      attnb, wob, b_out, d_out, SEQ, EMB, EMB);
}
